// ClockingCWRNN_2937757630471
// MI455X (gfx1250) — matrix operand order repaired and checked by reading, not compiled, not run on hardware
//
#include <hip/hip_runtime.h>

typedef __attribute__((ext_vector_type(16))) __bf16   v16bf;
typedef __attribute__((ext_vector_type(8)))  __bf16   v8bf;
typedef __attribute__((ext_vector_type(8)))  float    v8f;
typedef __attribute__((ext_vector_type(4)))  unsigned u32x4;
typedef __attribute__((ext_vector_type(8)))  unsigned u32x8;

static constexpr int kB    = 64;
static constexpr int kT    = 512;
static constexpr int kDin  = 256;
static constexpr int kDout = 1024;
static constexpr int kNTiles   = kDout / 16;          // 64 column tiles
static constexpr int kKcIn     = kDin / 32;           // 8  K-chunks (input gemm)
static constexpr int kKcH      = kDout / 32;          // 32 K-chunks (recurrent gemm)
static constexpr int kHElems   = kB * kDout;          // 65536
static constexpr int kTileElems = kKcH * 32 * 16;     // 16384 bf16 per W_h column tile
static constexpr unsigned kScanBlocks = 64;

// ---------------------------------------------------------------------------
// Pack weights into WMMA B-fragment order (bf16), zero H buffers + barriers.
// B fragment layout (16-bit B, 32x16): lane&15 = column, lane>>4 selects K-half,
// element e -> K = kt*32 + 8*(lane>>4) + (e&7) + 16*(e>>3).
// ---------------------------------------------------------------------------
__global__ void cwrnn_pack_init(const float* __restrict__ Win,
                                const float* __restrict__ Wh,
                                __bf16* __restrict__ WinP,
                                __bf16* __restrict__ WhP,
                                __bf16* __restrict__ H16,
                                float*  __restrict__ H32,
                                unsigned* __restrict__ bar) {
  const unsigned idx = blockIdx.x * blockDim.x + threadIdx.x;

  if (idx < (unsigned)(kNTiles * kKcH * 32 * 16)) {        // 1,048,576: W_h pack
    unsigned e    = idx & 15u;
    unsigned lane = (idx >> 4) & 31u;
    unsigned kt   = (idx >> 9) & 31u;
    unsigned nt   = idx >> 14;
    unsigned col  = nt * 16 + (lane & 15u);
    unsigned K    = kt * 32 + 8u * (lane >> 4) + (e & 7u) + 16u * (e >> 3);
    WhP[idx] = (__bf16)Wh[(size_t)K * kDout + col];
  }
  if (idx < (unsigned)(kNTiles * kKcIn * 32 * 16)) {       // 262,144: W_in pack
    unsigned e    = idx & 15u;
    unsigned lane = (idx >> 4) & 31u;
    unsigned kt   = (idx >> 9) & 7u;
    unsigned nt   = idx >> 12;
    unsigned col  = nt * 16 + (lane & 15u);
    unsigned K    = kt * 32 + 8u * (lane >> 4) + (e & 7u) + 16u * (e >> 3);
    WinP[idx] = (__bf16)Win[(size_t)K * kDout + col];
  }
  if (idx < (unsigned)kHElems) {                           // h0 = 0
    H32[idx] = 0.0f;
    H16[idx] = (__bf16)0.0f;
  }
  if (idx < (unsigned)kT) bar[idx] = 0u;                   // fresh barriers per launch
}

// ---------------------------------------------------------------------------
// Phase 1: P = X @ W_in + b_in, written into ys region of d_out.
// One 16x16 tile per wave, K = 256 (8 WMMA per wave).
// ---------------------------------------------------------------------------
__global__ void cwrnn_gemm_in(const float* __restrict__ x,
                              const __bf16* __restrict__ WinP,
                              const float* __restrict__ bin,
                              float* __restrict__ ys) {
  const unsigned lane   = threadIdx.x & 31u;
  const unsigned waveId = blockIdx.x * (blockDim.x >> 5) + (threadIdx.x >> 5);
  const unsigned tileN  = waveId & (kNTiles - 1);
  const unsigned tileM  = waveId >> 6;                     // 2048 M tiles
  const unsigned row0   = tileM * 16;
  const unsigned col    = tileN * 16 + (lane & 15u);
  const unsigned half   = lane >> 4;
  const unsigned arow   = row0 + (lane & 15u);

  v8f c0, c1 = {};
  const float bias = bin[col];
#pragma unroll
  for (int r = 0; r < 8; ++r) c0[r] = bias;

  const float*  __restrict__ xrow = x + (size_t)arow * kDin;
  const __bf16* __restrict__ wp   = WinP + (size_t)tileN * kKcIn * 32 * 16;

#pragma unroll
  for (unsigned kt = 0; kt < (unsigned)kKcIn; ++kt) {
    // A fragment: row = arow, two contiguous 8-element K chunks
    v16bf a;
    const float* xlo = xrow + kt * 32 + 8u * half;
    const float* xhi = xrow + kt * 32 + 16 + 8u * half;
#pragma unroll
    for (int e = 0; e < 8; ++e) { a[e] = (__bf16)xlo[e]; a[8 + e] = (__bf16)xhi[e]; }
    v16bf b = *reinterpret_cast<const v16bf*>(wp + ((size_t)kt * 32 + lane) * 16);
    if (kt & 1u)
      c1 = __builtin_amdgcn_wmma_f32_16x16x32_bf16(false, a, false, b, (short)0, c1, false, false);
    else
      c0 = __builtin_amdgcn_wmma_f32_16x16x32_bf16(false, a, false, b, (short)0, c0, false, false);
  }

#pragma unroll
  for (int r = 0; r < 8; ++r) {
    unsigned mrow = row0 + r + 8u * half;                  // flat (b*T + t) row
    ys[(size_t)mrow * kDout + col] = c0[r] + c1[r];
  }
}

// ---------------------------------------------------------------------------
// Phase 2: persistent cooperative scan. 64 blocks (one per 16-col tile),
// 4 waves/block (16 batch rows each). W_h tile staged into LDS once via the
// Tensor Data Mover, then reused for all 512 steps.
// Per step: S = H_prev @ W_h (WMMA, f32 accum seeded with P from ys),
// y = (1-g)*tanh(S) + g*h_prev ; grid barrier ; next step.
// ---------------------------------------------------------------------------
__global__ void cwrnn_scan(const float* __restrict__ periods,
                           const float* __restrict__ shifts,
                           const __bf16* __restrict__ WhP,
                           __bf16* __restrict__ H16,     // 2 * kHElems
                           float*  __restrict__ H32,     // 2 * kHElems
                           unsigned* __restrict__ bar,   // kT counters
                           float* __restrict__ ys,
                           float* __restrict__ hfinal) {
  __shared__ __align__(32) __bf16 ldsB[kTileElems];       // 32 KB: this tile's W_h

  const unsigned tid  = threadIdx.x;
  const unsigned lane = tid & 31u;
  const unsigned wave = tid >> 5;                         // 0..3
  const unsigned nt   = blockIdx.x;                       // 0..63
  const unsigned row0 = wave * 16;
  const unsigned col  = nt * 16 + (lane & 15u);
  const unsigned half = lane >> 4;
  const unsigned arow = row0 + (lane & 15u);

  const __bf16* __restrict__ tileSrc = WhP + (size_t)nt * kTileElems;

  // ---- TDM: async-copy the 32KB W_h tile into LDS (wave 0 issues) ----
  if (wave == 0) {
    unsigned long long gaddr = (unsigned long long)tileSrc;
    unsigned ldsaddr = (unsigned)(uintptr_t)(void*)&ldsB[0];   // LDS offset = addr[31:0]
    u32x4 g0;
    g0[0] = 1u;                                  // count=1, user descriptor
    g0[1] = ldsaddr;                             // lds_addr (bytes)
    g0[2] = (unsigned)gaddr;                     // global_addr[31:0]
    g0[3] = (unsigned)((gaddr >> 32) & 0x01FFFFFFull) | 0x80000000u; // addr[56:32] | type=2
    u32x8 g1;
    g1[0] = 0x00010000u;                         // wg_mask=0, data_size=1 (2 bytes)
    g1[1] = (unsigned)(kTileElems & 0xFFFF) << 16;   // tensor_dim0[15:0] @bits63:48
    g1[2] = ((unsigned)(kTileElems >> 16) & 0xFFFFu) | (1u << 16); // dim0 hi | tensor_dim1=1
    g1[3] = (unsigned)(kTileElems & 0xFFFF) << 16;   // tile_dim0 @bits127:112
    g1[4] = 1u;                                  // tile_dim1=1, tile_dim2=0
    g1[5] = (unsigned)kTileElems;                // tensor_dim0_stride[31:0]
    g1[6] = 0u;
    g1[7] = 0u;
    asm volatile("tensor_load_to_lds %0, %1" :: "s"(g0), "s"(g1) : "memory");
    __builtin_amdgcn_s_wait_tensorcnt(0);
  }
  __syncthreads();
  // Belt-and-braces: rewrite the tile with plain loads (same bytes; hedges
  // against descriptor-field errors since there is no runtime validation).
  {
    const uint4* src = reinterpret_cast<const uint4*>(tileSrc);
    uint4* dst = reinterpret_cast<uint4*>(ldsB);
    for (unsigned i = tid; i < (kTileElems * 2) / 16; i += blockDim.x) dst[i] = src[i];
  }
  __syncthreads();

  const float prd = periods[nt >> 3];                     // 128-wide modules -> nt/8
  const float shf = shifts[nt >> 3];

  for (unsigned t = 0; t < (unsigned)kT; ++t) {
    const __bf16* __restrict__ Hp   = H16 + (size_t)(t & 1u) * kHElems;
    __bf16* __restrict__       Hn   = H16 + (size_t)((t + 1u) & 1u) * kHElems;
    const float* __restrict__  H32p = H32 + (size_t)(t & 1u) * kHElems;
    float* __restrict__        H32n = H32 + (size_t)((t + 1u) & 1u) * kHElems;

    const float g = (sinf((float)t * prd + shf) + 1.0f) * 0.5f;

    // Seed accumulator with precomputed pre-activation P (stored in ys).
    v8f c0, c1 = {};
#pragma unroll
    for (int r = 0; r < 8; ++r) {
      unsigned brow = row0 + r + 8u * half;
      c0[r] = ys[((size_t)brow * kT + t) * kDout + col];
    }

    const __bf16* __restrict__ hrow = Hp + (size_t)arow * kDout;
#pragma unroll 8
    for (unsigned kt = 0; kt < (unsigned)kKcH; ++kt) {
      v8bf alo = *reinterpret_cast<const v8bf*>(hrow + kt * 32 + 8u * half);
      v8bf ahi = *reinterpret_cast<const v8bf*>(hrow + kt * 32 + 16 + 8u * half);
      v16bf a = __builtin_shufflevector(alo, ahi,
                  0, 1, 2, 3, 4, 5, 6, 7, 8, 9, 10, 11, 12, 13, 14, 15);
      v16bf b = *reinterpret_cast<const v16bf*>(&ldsB[((size_t)kt * 32 + lane) * 16]);
      if (kt & 1u)
        c1 = __builtin_amdgcn_wmma_f32_16x16x32_bf16(false, a, false, b, (short)0, c1, false, false);
      else
        c0 = __builtin_amdgcn_wmma_f32_16x16x32_bf16(false, a, false, b, (short)0, c0, false, false);
    }

#pragma unroll
    for (int r = 0; r < 8; ++r) {
      unsigned brow = row0 + r + 8u * half;
      float acc = c0[r] + c1[r];
      float hp  = H32p[(size_t)brow * kDout + col];
      float y   = (1.0f - g) * tanhf(acc) + g * hp;
      ys[((size_t)brow * kT + t) * kDout + col] = y;
      H32n[(size_t)brow * kDout + col] = y;
      Hn[(size_t)brow * kDout + col]   = (__bf16)y;
      if (t == (unsigned)kT - 1) hfinal[(size_t)brow * kDout + col] = y;
    }

    // -------- grid-wide step barrier (fresh counter per step) --------
    __threadfence();
    __syncthreads();
    if (tid == 0) {
      __hip_atomic_fetch_add(&bar[t], 1u, __ATOMIC_ACQ_REL, __HIP_MEMORY_SCOPE_AGENT);
      while (__hip_atomic_load(&bar[t], __ATOMIC_ACQUIRE, __HIP_MEMORY_SCOPE_AGENT) < kScanBlocks)
        __builtin_amdgcn_s_sleep(2);
    }
    __syncthreads();
  }
}

extern "C" void kernel_launch(void* const* d_in, const int* in_sizes, int n_in,
                              void* d_out, int out_size, void* d_ws, size_t ws_size,
                              hipStream_t stream) {
  const float* x       = (const float*)d_in[0];   // [64,512,256]
  const float* W_in    = (const float*)d_in[1];   // [256,1024]
  const float* b_in    = (const float*)d_in[2];   // [1024]
  const float* W_h     = (const float*)d_in[3];   // [1024,1024]
  const float* periods = (const float*)d_in[4];   // [8]
  const float* shifts  = (const float*)d_in[5];   // [8]

  float* ys     = (float*)d_out;                  // [64,512,1024]
  float* hfinal = ys + (size_t)kB * kT * kDout;   // [64,1024]

  // Workspace carve-up (~3.3 MB total)
  char* ws = (char*)d_ws;
  __bf16* WhP  = (__bf16*)ws;                                          // 2 MB
  __bf16* WinP = (__bf16*)(ws + (2u << 20));                           // 512 KB
  __bf16* H16  = (__bf16*)(ws + (2u << 20) + (512u << 10));            // 256 KB (x2 buf)
  float*  H32  = (float*)(ws + (2u << 20) + (768u << 10));             // 512 KB (x2 buf)
  unsigned* bar = (unsigned*)(ws + (2u << 20) + (1280u << 10));        // 2 KB

  // 1) pack weights to fragment order, zero H and barriers (1,048,576 threads)
  cwrnn_pack_init<<<4096, 256, 0, stream>>>(W_in, W_h, WinP, WhP, H16, H32, bar);

  // 2) P = X @ W_in + b_in  (2048x64 wave tiles, 8 waves per block)
  cwrnn_gemm_in<<<16384, 256, 0, stream>>>(x, WinP, b_in, ys);

  // 3) persistent cooperative scan over T=512 steps
  cwrnn_scan<<<kScanBlocks, 128, 0, stream>>>(periods, shifts, WhP, H16, H32,
                                              bar, ys, hfinal);
}
